// InputMaskingNet_42949672960343
// MI455X (gfx1250) — compile-verified, matrix operand order checked by reading, not run on hardware
//
#include <hip/hip_runtime.h>
#include <hip/hip_bf16.h>

// ---------------------------------------------------------------------------
// Shapes / constants from the reference
// ---------------------------------------------------------------------------
#define DIM      512
#define IN_LEN   16384
#define OUT_LEN  1024
#define STRIDE   16
#define BATCH    8
#define MROWS    (OUT_LEN * BATCH)   // 8192 GEMM rows
#define JDIM     (BATCH * DIM)       // 4096 (b,dim) chains

typedef __attribute__((ext_vector_type(16))) __bf16 v16bf;
typedef __attribute__((ext_vector_type(8)))  __bf16 v8bf;
typedef __attribute__((ext_vector_type(8)))  float  v8f;

struct bf16x4 { __bf16 v[4]; };

// ---------------------------------------------------------------------------
// 1) Downsample x[::16] and convert to bf16 (row-major 8192 x 512)
// ---------------------------------------------------------------------------
__global__ void downsample_bf16_kernel(const float* __restrict__ x,
                                       __bf16* __restrict__ xd)
{
    const size_t i    = (size_t)blockIdx.x * blockDim.x + threadIdx.x;
    const size_t base = i * 4;                 // element index in (1024*4096)
    const int t = (int)(base >> 12);           // row (t in 0..1023; row = t*8+b folded)
    const int j = (int)(base & 4095);
    // xd row m corresponds to x row m of the strided view: x[(t*16)] over (B*DIM)
    const float4 v = *(const float4*)(x + ((size_t)t * STRIDE) * JDIM + j);
    bf16x4 o;
    o.v[0] = (__bf16)v.x; o.v[1] = (__bf16)v.y;
    o.v[2] = (__bf16)v.z; o.v[3] = (__bf16)v.w;
    *(bf16x4*)(xd + base) = o;
}

// ---------------------------------------------------------------------------
// 2) Convert both 512x512 weight matrices to bf16
// ---------------------------------------------------------------------------
__global__ void weights_bf16_kernel(const float* __restrict__ w1,
                                    const float* __restrict__ w2,
                                    __bf16* __restrict__ w1b,
                                    __bf16* __restrict__ w2b)
{
    const size_t i    = (size_t)blockIdx.x * blockDim.x + threadIdx.x;
    const size_t base = i * 4;                 // < 512*512
    const float4 a = *(const float4*)(w1 + base);
    const float4 b = *(const float4*)(w2 + base);
    bf16x4 oa, ob;
    oa.v[0] = (__bf16)a.x; oa.v[1] = (__bf16)a.y; oa.v[2] = (__bf16)a.z; oa.v[3] = (__bf16)a.w;
    ob.v[0] = (__bf16)b.x; ob.v[1] = (__bf16)b.y; ob.v[2] = (__bf16)b.z; ob.v[3] = (__bf16)b.w;
    *(bf16x4*)(w1b + base) = oa;
    *(bf16x4*)(w2b + base) = ob;
}

// ---------------------------------------------------------------------------
// 3) WMMA GEMM:  out[m][n] = sum_k A[m][k] * W[n][k] + bias[n]
//    A: (M x 512) bf16 row-major, W: (512 x 512) bf16 row-major.
//    One 16x16 output tile per wave, K walked in steps of 32 via
//    v_wmma_f32_16x16x32_bf16.
//
//    A-matrix lane layout (ISA 7.12.2, 16-bit A 16x32):
//      lane L (L<16)  : row M=L,  K = {kk+0..7,  kk+16..23}
//      lane L (L>=16) : row M=L-16, K = {kk+8..15, kk+24..31}
//    -> two contiguous 16-byte loads per lane per K-step.
//    B (K x 16) per-column layout mirrors this; column n of B is row n of W
//    (contiguous in k), so B uses the identical addressing off W's row n.
// ---------------------------------------------------------------------------
__global__ void gemm_bf16_wmma_kernel(const __bf16* __restrict__ A,
                                      const __bf16* __restrict__ W,
                                      const float*  __restrict__ bias,
                                      float*  __restrict__ outF,
                                      __bf16* __restrict__ outB,
                                      int storeBf16, int M)
{
    constexpr int K = DIM;
    constexpr int N = DIM;
    constexpr int tilesN = N / 16;            // 32

    const int wave = threadIdx.x >> 5;
    const int lane = threadIdx.x & 31;
    const int tile = blockIdx.x * (blockDim.x >> 5) + wave;
    const int tm   = tile / tilesN;
    const int tn   = tile % tilesN;
    if (tm * 16 >= M) return;                 // uniform per wave -> EXEC stays all-ones

    const int half = lane >> 4;               // 0 or 1
    const int idx  = lane & 15;

    const __bf16* arow = A + (size_t)(tm * 16 + idx) * K;
    const __bf16* wrow = W + (size_t)(tn * 16 + idx) * K;

    v8f c = {};
#pragma unroll
    for (int kk = 0; kk < K; kk += 32) {
        const v8bf a0 = *(const v8bf*)(arow + kk + half * 8);
        const v8bf a1 = *(const v8bf*)(arow + kk + 16 + half * 8);
        const v8bf b0 = *(const v8bf*)(wrow + kk + half * 8);
        const v8bf b1 = *(const v8bf*)(wrow + kk + 16 + half * 8);
        v16bf a, b;
#pragma unroll
        for (int i = 0; i < 8; ++i) {
            a[i] = a0[i]; a[i + 8] = a1[i];
            b[i] = b0[i]; b[i + 8] = b1[i];
        }
        // (neg_a, A, neg_b, B, c_mod, C, reuse_a, reuse_b)
        c = __builtin_amdgcn_wmma_f32_16x16x32_bf16(
                false, a, false, b, (short)0, c, false, false);
    }

    // C/D layout: lane holds column n = tn*16+idx; VGPR r holds row tm*16 + r + 8*half
    const int   n  = tn * 16 + idx;
    const float bn = bias[n];
#pragma unroll
    for (int r = 0; r < 8; ++r) {
        const int   m   = tm * 16 + r + half * 8;
        const float val = c[r] + bn;
        if (storeBf16) outB[(size_t)m * N + n] = (__bf16)val;
        else           outF[(size_t)m * N + n] = val;
    }
}

// ---------------------------------------------------------------------------
// 4) LIF scan over T=1024 for each of 4096 (b,dim) chains.
//    v = (v + x)/2 ; spike = (v - 1 >= 0) ; v *= (1 - spike)
// ---------------------------------------------------------------------------
__global__ void lif_scan_kernel(const float* __restrict__ h,
                                float* __restrict__ s)
{
    const int j = blockIdx.x * blockDim.x + threadIdx.x;   // 0..4095
    float v = 0.0f;
#pragma unroll 4
    for (int t = 0; t < OUT_LEN; ++t) {
        const float x  = h[(size_t)t * JDIM + j];
        v = 0.5f * (v + x);
        const float sp = (v - 1.0f >= 0.0f) ? 1.0f : 0.0f;
        s[(size_t)t * JDIM + j] = sp;
        v = (sp != 0.0f) ? 0.0f : v;
    }
}

// ---------------------------------------------------------------------------
// 5) Linear upsample 1024 -> 16384 along time, float4-vectorized.
//    out[l] = s[lo]*(1-w) + s[hi]*w,  pos = l*(T-1)/(L-1)
// ---------------------------------------------------------------------------
__global__ void upsample_kernel(const float* __restrict__ s,
                                float* __restrict__ out)
{
    const size_t i    = (size_t)blockIdx.x * blockDim.x + threadIdx.x;
    const size_t base = i * 4;                  // element index in (16384*4096)
    const int l = (int)(base >> 12);
    const int j = (int)(base & 4095);

    const float pos = (float)l * ((float)(OUT_LEN - 1) / (float)(IN_LEN - 1));
    int lo = (int)pos;                          // pos >= 0 -> trunc == floor
    const float w = pos - (float)lo;
    int hi = lo + 1; if (hi > OUT_LEN - 1) hi = OUT_LEN - 1;

    const float4 a = *(const float4*)(s + (size_t)lo * JDIM + j);
    const float4 b = *(const float4*)(s + (size_t)hi * JDIM + j);
    float4 o;
    o.x = a.x + (b.x - a.x) * w;
    o.y = a.y + (b.y - a.y) * w;
    o.z = a.z + (b.z - a.z) * w;
    o.w = a.w + (b.w - a.w) * w;
    *(float4*)(out + base) = o;
}

// ---------------------------------------------------------------------------
// Launch
// ---------------------------------------------------------------------------
extern "C" void kernel_launch(void* const* d_in, const int* in_sizes, int n_in,
                              void* d_out, int out_size, void* d_ws, size_t ws_size,
                              hipStream_t stream)
{
    const float* x      = (const float*)d_in[0];   // (16384, 8, 512)
    const float* conv_w = (const float*)d_in[1];   // (512, 512, 1)
    const float* conv_b = (const float*)d_in[2];   // (512,)
    const float* lin_w  = (const float*)d_in[3];   // (512, 512)
    const float* lin_b  = (const float*)d_in[4];   // (512,)
    float* out = (float*)d_out;                    // (16384, 8, 512)

    // Workspace layout (bytes): all regions are written before read each call.
    char* ws = (char*)d_ws;
    __bf16* xd   = (__bf16*)(ws);                              //  8 MiB: 8192*512 bf16
    __bf16* w1b  = (__bf16*)(ws + ((size_t)8  << 20));         //  512 KiB
    __bf16* w2b  = w1b + (size_t)DIM * DIM;                    //  512 KiB
    __bf16* h1   = (__bf16*)(ws + ((size_t)9  << 20));         //  8 MiB: 8192*512 bf16
    float*  h    = (float*) (ws + ((size_t)17 << 20));         // 16 MiB: 8192*512 f32
    float*  sp   = (float*) (ws + ((size_t)34 << 20));         // 16 MiB: 8192*512 f32

    // 1) downsample + bf16 convert: 1024*4096/4 threads
    downsample_bf16_kernel<<<(OUT_LEN * JDIM / 4) / 256, 256, 0, stream>>>(x, xd);

    // 2) weights -> bf16: 512*512/4 threads
    weights_bf16_kernel<<<(DIM * DIM / 4) / 256, 256, 0, stream>>>(conv_w, lin_w, w1b, w2b);

    // 3) GEMM1 (store bf16 for GEMM2 input): 512x32 tiles, 8 waves/block
    const int nTiles  = (MROWS / 16) * (DIM / 16);   // 16384
    const int nBlocks = nTiles / 8;                  // 2048
    gemm_bf16_wmma_kernel<<<nBlocks, 256, 0, stream>>>(xd,  w1b, conv_b, nullptr, h1, 1, MROWS);

    // 4) GEMM2 (store f32 membrane input)
    gemm_bf16_wmma_kernel<<<nBlocks, 256, 0, stream>>>(h1, w2b, lin_b, h, nullptr, 0, MROWS);

    // 5) LIF scan: 4096 chains
    lif_scan_kernel<<<JDIM / 256, 256, 0, stream>>>(h, sp);

    // 6) upsample to 16384 rows: 16384*4096/4 threads
    upsample_kernel<<<(size_t)(IN_LEN) * JDIM / 4 / 256, 256, 0, stream>>>(sp, out);
}
